// S6_43069932044292
// MI455X (gfx1250) — matrix operand order repaired and checked by reading, not compiled, not run on hardware
//
#include <hip/hip_runtime.h>

typedef __attribute__((ext_vector_type(16))) __bf16 v16bf;
typedef __attribute__((ext_vector_type(8)))  __bf16 v8bf;
typedef __attribute__((ext_vector_type(8)))  float  v8f;

#define BATCH   4
#define SEQ     2048
#define DMODEL  1024
#define NSTATE  16
#define MROWS   (BATCH*SEQ)
#define NW1     (DMODEL*DMODEL)   /* 1048576 */
#define NW2     (NSTATE*DMODEL)   /* 16384   */

/* workspace layout, in u16 elements (bf16 planes) */
#define WS_W1H  0
#define WS_W1L  (NW1)
#define WS_W2H  (2*NW1)
#define WS_W2L  (2*NW1 + NW2)
#define WS_W3H  (2*NW1 + 2*NW2)
#define WS_W3L  (2*NW1 + 3*NW2)

/* LDS layout, bytes. x rows padded to 1032 bf16 (2064 B):
   row stride mod 256 B == 16 -> each of 16 lanes' b128 reads hits a
   distinct bank quad (all 64 banks covered per half-wave). */
#define RSTRIDE   2064
#define LDS_XH    0
#define LDS_XL    (16*RSTRIDE)              /* 33024 */
#define LDS_PART  (2*16*RSTRIDE)            /* 66048 */
#define LDS_PROD  (LDS_PART + 8*256*4)      /* 74240 */
#define LDS_BCS   (LDS_PROD + 256*4)        /* 75264 */
#define LDS_TOTAL (LDS_BCS + 64)            /* 75328 */

__device__ __forceinline__ unsigned f2bf(float f) {
  unsigned u = __float_as_uint(f);                    /* RNE bf16 truncation */
  return ((u + 0x7FFFu + ((u >> 16) & 1u)) >> 16) & 0xFFFFu;
}
__device__ __forceinline__ float bf2f(unsigned h) {
  return __uint_as_float(h << 16);
}

/* -------- split weights into bf16 hi/lo planes (one cheap pass) -------- */
__global__ __launch_bounds__(256) void S6_split_w(
    const float* __restrict__ W1, const float* __restrict__ W2,
    const float* __restrict__ W3, unsigned short* __restrict__ ws) {
  int i = blockIdx.x * 256 + threadIdx.x;
  if (i < NW1) {
    float f = W1[i];
    unsigned h = f2bf(f);
    ws[WS_W1H + i] = (unsigned short)h;
    ws[WS_W1L + i] = (unsigned short)f2bf(f - bf2f(h));
  } else {
    int j = i - NW1;
    if (j < NW2) {
      float f = W2[j];
      unsigned h = f2bf(f);
      ws[WS_W2H + j] = (unsigned short)h;
      ws[WS_W2L + j] = (unsigned short)f2bf(f - bf2f(h));
      float g = W3[j];
      unsigned gh = f2bf(g);
      ws[WS_W3H + j] = (unsigned short)gh;
      ws[WS_W3L + j] = (unsigned short)f2bf(g - bf2f(gh));
    }
  }
}

/* A-fragment (16x32 bf16): lane l16 = row; lanes 0-15 take
   K 0..15, lanes 16-31 take K 16..31 -> two b128 LDS reads. */
__device__ __forceinline__ v16bf load_afrag(const char* xs, int ktbyte,
                                            int l16, int lhalf) {
  const char* p = xs + l16 * RSTRIDE + ktbyte + lhalf * 32;
  v16bf a;
  ((v8bf*)&a)[0] = *(const v8bf*)(p);
  ((v8bf*)&a)[1] = *(const v8bf*)(p + 16);
  return a;
}

/* B-fragment (32x16 bf16): lane l16 = column n (= weight row), lanes 0-15
   hold K 0..15, lanes 16-31 hold K 16..31 -> 32 contiguous bytes/lane. */
__device__ __forceinline__ v16bf load_bfrag(const unsigned short* wrow,
                                            int kbase, int lhalf) {
  return *(const v16bf*)(wrow + kbase + lhalf * 16);
}

/* bf16x3 compensated accumulate: acc += xh*wh + xh*wl + xl*wh */
__device__ __forceinline__ v8f wmma3(v8f acc, v16bf ah, v16bf al,
                                     v16bf bh, v16bf bl) {
  acc = __builtin_amdgcn_wmma_f32_16x16x32_bf16(false, ah, false, bh,
                                                (short)0, acc, false, false);
  acc = __builtin_amdgcn_wmma_f32_16x16x32_bf16(false, ah, false, bl,
                                                (short)0, acc, false, false);
  acc = __builtin_amdgcn_wmma_f32_16x16x32_bf16(false, al, false, bh,
                                                (short)0, acc, false, false);
  return acc;
}

__global__ __launch_bounds__(256) void S6_fused(
    const float* __restrict__ xg, const float* __restrict__ b1,
    const float* __restrict__ b2, const float* __restrict__ b3,
    const unsigned short* __restrict__ wsu, float* __restrict__ yg) {
  extern __shared__ char smem[];
  const int m0    = blockIdx.x * 16;
  const int t     = threadIdx.x;
  const int w     = t >> 5;        /* wave 0..7 */
  const int lane  = t & 31;
  const int l16   = lane & 15;
  const int lhalf = lane >> 4;

  /* ---- Phase 1: stage x tile, split into bf16 hi/lo in LDS ---- */
  {
#pragma unroll
    for (int row = 0; row < 16; ++row) {
      const float4 v =
          ((const float4*)(xg + (size_t)(m0 + row) * DMODEL))[t];
      unsigned h0 = f2bf(v.x), h1 = f2bf(v.y), h2 = f2bf(v.z), h3 = f2bf(v.w);
      unsigned l0 = f2bf(v.x - bf2f(h0)), l1 = f2bf(v.y - bf2f(h1));
      unsigned l2 = f2bf(v.z - bf2f(h2)), l3 = f2bf(v.w - bf2f(h3));
      uint2 hv, lv;
      hv.x = h0 | (h1 << 16); hv.y = h2 | (h3 << 16);
      lv.x = l0 | (l1 << 16); lv.y = l2 | (l3 << 16);
      *(uint2*)(smem + LDS_XH + row * RSTRIDE + (t << 3)) = hv;
      *(uint2*)(smem + LDS_XL + row * RSTRIDE + (t << 3)) = lv;
    }
  }
  __syncthreads();

  const char* xsh = smem + LDS_XH;
  const char* xsl = smem + LDS_XL;

  /* ---- Phase 2: B = x@W2^T, C = x@W3^T partial tiles via WMMA ---- */
  {
    const unsigned short* Wh = wsu + (w < 4 ? WS_W2H : WS_W3H);
    const unsigned short* wrowh = Wh + (size_t)l16 * DMODEL;
    const int kt0 = (w & 3) * 8;
    v8f acc = {};
#pragma unroll
    for (int kk = 0; kk < 8; ++kk) {
      const int kt = kt0 + kk;
      v16bf ah = load_afrag(xsh, kt * 64, l16, lhalf);
      v16bf al = load_afrag(xsl, kt * 64, l16, lhalf);
      v16bf bh = load_bfrag(wrowh, kt * 32, lhalf);
      v16bf bl = load_bfrag(wrowh + NW2, kt * 32, lhalf); /* lo plane */
      acc = wmma3(acc, ah, al, bh, bl);
    }
    float* part = (float*)(smem + LDS_PART) + w * 256;
#pragma unroll
    for (int v = 0; v < 8; ++v) {
      const int m = v + 8 * lhalf;           /* C/D layout: M=v(+8), N=l16 */
      part[m * 16 + l16] = acc[v];
    }
  }
  __syncthreads();

  /* reduce partials: (B+b2)*(C+b3), then row-sum -> bc[m] */
  {
    const int m = t >> 4, n = t & 15;
    const float* part = (const float*)(smem + LDS_PART);
    float bs = 0.f, cs = 0.f;
#pragma unroll
    for (int p = 0; p < 4; ++p) {
      bs += part[p * 256 + m * 16 + n];
      cs += part[(p + 4) * 256 + m * 16 + n];
    }
    bs += b2[n]; cs += b3[n];
    ((float*)(smem + LDS_PROD))[m * 16 + n] = bs * cs;
  }
  __syncthreads();
  if (t < 16) {
    const float* prod = (const float*)(smem + LDS_PROD) + t * 16;
    float s = 0.f;
#pragma unroll
    for (int n = 0; n < 16; ++n) s += prod[n];
    ((float*)(smem + LDS_BCS))[t] = s;
  }
  __syncthreads();

  /* ---- Phase 3: delta = x@W1^T, fused softplus / scale epilogue ----
     Wave w owns columns [w*128, w*128+128): 8 N-tiles, K-outer loop so
     each A-fragment pair feeds 24 WMMAs. */
  float bcv[8];
  {
    const float* bcs = (const float*)(smem + LDS_BCS);
#pragma unroll
    for (int v = 0; v < 8; ++v) bcv[v] = bcs[v + 8 * lhalf];
  }
  const unsigned short* W1h = wsu + WS_W1H;
  v8f acc[8];
#pragma unroll
  for (int i = 0; i < 8; ++i) acc[i] = (v8f){};

  for (int kt = 0; kt < 32; ++kt) {
    v16bf ah = load_afrag(xsh, kt * 64, l16, lhalf);
    v16bf al = load_afrag(xsl, kt * 64, l16, lhalf);
#pragma unroll
    for (int i = 0; i < 8; ++i) {
      const unsigned short* wrowh =
          W1h + (size_t)(w * 128 + i * 16 + l16) * DMODEL;
      v16bf bh = load_bfrag(wrowh, kt * 32, lhalf);
      v16bf bl = load_bfrag(wrowh + NW1, kt * 32, lhalf); /* W1 lo plane */
      acc[i] = wmma3(acc[i], ah, al, bh, bl);
    }
  }

#pragma unroll
  for (int i = 0; i < 8; ++i) {
    const int n0 = w * 128 + i * 16;
    const float b1v = b1[n0 + l16];
#pragma unroll
    for (int v = 0; v < 8; ++v) {
      const int m = v + 8 * lhalf;
      const float pre = acc[i][v] + b1v;
      /* stable softplus: max(z,0) + log1p(exp(-|z|)) */
      const float sp = fmaxf(pre, 0.f) + log1pf(expf(-fabsf(pre)));
      const size_t g = (size_t)(m0 + m) * DMODEL + (n0 + l16);
      yg[g] = xg[g] * sp * bcv[v];
    }
  }
}

extern "C" void kernel_launch(void* const* d_in, const int* in_sizes, int n_in,
                              void* d_out, int out_size, void* d_ws,
                              size_t ws_size, hipStream_t stream) {
  const float* x  = (const float*)d_in[0];
  const float* W1 = (const float*)d_in[1];
  const float* b1 = (const float*)d_in[2];
  const float* W2 = (const float*)d_in[3];
  const float* b2 = (const float*)d_in[4];
  const float* W3 = (const float*)d_in[5];
  const float* b3 = (const float*)d_in[6];
  /* d_in[7] = A : mathematically dead in the reference */
  unsigned short* ws = (unsigned short*)d_ws;
  float* y = (float*)d_out;

  const int splitBlocks = (NW1 + NW2 + 255) / 256;
  S6_split_w<<<splitBlocks, 256, 0, stream>>>(W1, W2, W3, ws);
  S6_fused<<<MROWS / 16, 256, LDS_TOTAL, stream>>>(x, b1, b2, b3, ws, y);
}
